// HidsModel_53807350284864
// MI455X (gfx1250) — matrix operand order repaired and checked by reading, not compiled, not run on hardware
//
#include <hip/hip_runtime.h>

// Problem dims (fixed by the reference)
#define BB  64
#define TT  256
#define DD  256
#define HH  256
#define LL  50
#define OO  2
#define G4H 1024   // 4*H

#define LDP  40    // padded row length (ushort) for 32-wide LDS tiles
#define WLP  264   // padded row length (ushort) for 256-wide Whh LDS rows

typedef __attribute__((ext_vector_type(16))) __bf16 v16bf;
typedef __attribute__((ext_vector_type(8)))  float  v8f;
typedef __attribute__((__vector_size__(16))) int    v4i;

#define GAS __attribute__((address_space(1)))
#define LAS __attribute__((address_space(3)))

union Frag16 { v16bf v; uint4 u[2]; };

#if __has_builtin(__builtin_amdgcn_global_load_async_to_lds_b128)
#define HAVE_ASYNC_LDS 1
#else
#define HAVE_ASYNC_LDS 0
#endif

__device__ __forceinline__ void async_copy16(unsigned short* lds,
                                             const unsigned short* g) {
#if HAVE_ASYNC_LDS
  __builtin_amdgcn_global_load_async_to_lds_b128((GAS v4i*)g, (LAS v4i*)lds, 0, 0);
#else
  *(uint4*)lds = *(const uint4*)g;
#endif
}
__device__ __forceinline__ void wait_async() {
#if HAVE_ASYNC_LDS
#if __has_builtin(__builtin_amdgcn_s_wait_asynccnt)
  __builtin_amdgcn_s_wait_asynccnt(0);
#else
  asm volatile("s_wait_asynccnt 0" ::: "memory");
#endif
#endif
}

__device__ __forceinline__ unsigned short f2bf(float x) {
  union { float f; unsigned u; } c; c.f = x;
  unsigned r = c.u + 0x7fffu + ((c.u >> 16) & 1u);   // round-to-nearest-even
  return (unsigned short)(r >> 16);
}
__device__ __forceinline__ float bf2f(unsigned short b) {
  union { unsigned u; float f; } c; c.u = ((unsigned)b) << 16;
  return c.f;
}
__device__ __forceinline__ float sigm(float x) { return 1.f / (1.f + __expf(-x)); }

// ---------------------------------------------------------------------------
// f32 -> bf16 conversion (grid-stride)
// ---------------------------------------------------------------------------
__global__ void k_f32_to_bf16(const float* __restrict__ s,
                              unsigned short* __restrict__ d, long n) {
  long i  = (long)blockIdx.x * blockDim.x + threadIdx.x;
  long st = (long)gridDim.x * blockDim.x;
  for (; i < n; i += st) d[i] = f2bf(s[i]);
}

// Copy a 64x32 bf16 tile (global row stride HH) into a padded LDS tile
// (row stride LDP) via the async direct-to-LDS DMA path.  128 threads,
// each copies 2x16B.
__device__ __forceinline__ void stage64x32(unsigned short* lds,
                                           const unsigned short* g, int tid) {
  #pragma unroll
  for (int i = 0; i < 2; ++i) {
    const int sid = tid + i * 128;
    const int row = sid >> 2, sg = (sid & 3) * 8;
    async_copy16(lds + row * LDP + sg, g + (long)row * HH + sg);
  }
}

// ---------------------------------------------------------------------------
// Input projection GEMM: xg[M,4H] = X[M,K] * W[4H,K]^T + (bih + bhh)
// M = B*T = 16384, K = H = 256, N = 4H = 1024.  bf16 in, f32 accumulate.
// Grid (N/64, M/64), block 128 (4 waves).  K-chunks of 32 are double-
// buffered through LDS with async-to-LDS DMA: chunk k+1 streams in while
// chunk k's fragments are read with ds_load_b128 and fed to v_wmma.
// ---------------------------------------------------------------------------
__global__ __launch_bounds__(128) void k_xproj(
    const unsigned short* __restrict__ X,    // [M, K] bf16 (row-major)
    const unsigned short* __restrict__ W,    // [4H, K] bf16 (layer slice)
    const float* __restrict__ bih,           // [4H]
    const float* __restrict__ bhh,           // [4H]
    float* __restrict__ xg)                  // [M, 4H]
{
  __shared__ unsigned short Xs[2][64][LDP];  // ~10 KB (padded vs bank conflicts)
  __shared__ unsigned short Ws[2][64][LDP];  // ~10 KB

  const int tid  = threadIdx.x;
  const int lane = tid & 31;
  const int wv   = tid >> 5;
  const int m0   = blockIdx.y * 64;
  const int n0   = blockIdx.x * 64;
  const int half = lane >> 4;       // 0: lanes 0-15, 1: lanes 16-31
  const int lr   = lane & 15;

  const unsigned short* Xb = X + (long)m0 * HH;
  const unsigned short* Wb = W + (long)n0 * HH;

  v8f acc[4] = {v8f{}, v8f{}, v8f{}, v8f{}};

  stage64x32(&Xs[0][0][0], Xb, tid);
  stage64x32(&Ws[0][0][0], Wb, tid);
  wait_async();
  __syncthreads();

  #pragma unroll
  for (int kc = 0; kc < 8; ++kc) {
    const int cur = kc & 1, nxt = cur ^ 1;
    if (kc < 7) {                            // DMA next chunk during WMMAs
      const int k1 = (kc + 1) * 32;
      stage64x32(&Xs[nxt][0][0], Xb + k1, tid);
      stage64x32(&Ws[nxt][0][0], Wb + k1, tid);
    }
    // A fragment: lanes 0-15 K=+0..7/+16..23, lanes 16-31 K=+8..15/+24..31.
    Frag16 a;
    a.u[0] = *(const uint4*)&Xs[cur][wv * 16 + lr][half * 8];
    a.u[1] = *(const uint4*)&Xs[cur][wv * 16 + lr][16 + half * 8];
    #pragma unroll
    for (int nt = 0; nt < 4; ++nt) {
      // B fragment: lanes 0-15 K=+0..7/+16..23, lanes 16-31 K=+8..15/+24..31 of column nt*16+lr.
      Frag16 bfr;
      bfr.u[0] = *(const uint4*)&Ws[cur][nt * 16 + lr][half * 8];
      bfr.u[1] = *(const uint4*)&Ws[cur][nt * 16 + lr][16 + half * 8];
      acc[nt] = __builtin_amdgcn_wmma_f32_16x16x32_bf16(
          false, a.v, false, bfr.v, (short)0, acc[nt], false, false);
    }
    wait_async();                            // next chunk's DMA landed
    __syncthreads();                         // everyone done with 'cur'
  }

  // D layout: VGPR v holds M = v + 8*half, N = lane%16.
  #pragma unroll
  for (int nt = 0; nt < 4; ++nt) {
    const int n = n0 + nt * 16 + lr;
    const float bias = bih[n] + bhh[n];
    #pragma unroll
    for (int v = 0; v < 8; ++v) {
      const int m = m0 + wv * 16 + v + half * 8;
      xg[(long)m * G4H + n] = acc[nt][v] + bias;
    }
  }
}

// ---------------------------------------------------------------------------
// Persistent recurrent scan for one layer.
// Grid: 64 WGs (4 batch tiles x 16 hidden tiles), block 128 (4 waves).
// Wave g computes gate g's 16x16 tile:  gates = h(t-1) @ Whh_gᵀ + xg(t).
// Whh slice (4 x 16 x 256 bf16) is DMA'd into (bank-conflict-padded) LDS
// once and reused for all 256 timesteps.  Cross-WG sync is per batch-group
// producer/consumer flags: flags[bm][t] counts hidden-tile producers (16).
// ---------------------------------------------------------------------------
__global__ __launch_bounds__(128) void k_scan(
    const float* __restrict__ xg,            // [M, 4H]
    const unsigned short* __restrict__ Whh,  // [4H, H] bf16 (layer slice)
    unsigned short* __restrict__ hseq,       // [B, T, H] bf16 (out; t-1 read)
    float* __restrict__ cbuf,                // [B, H] cell state
    int* __restrict__ flags)                 // [4][TT] producer counters
{
  __shared__ unsigned short Wl[4][16][WLP];  // ~33 KB (padded rows)
  __shared__ float gbuf[4][16][16];          //  4 KB

  const int lane = threadIdx.x & 31;
  const int g    = threadIdx.x >> 5;         // gate 0..3 (i,f,g,o)
  const int bm   = blockIdx.x & 3;           // batch tile
  const int hn   = blockIdx.x >> 2;          // hidden tile
  const int half = lane >> 4;
  const int lr   = lane & 15;

  // Stage this WG's Whh slice via async DMA: wave g loads its 16 rows x 256 K.
  {
    const int kb = half * 128;               // each lane-half copies 128 ushorts
    const unsigned short* src =
        Whh + (long)(g * HH + hn * 16 + lr) * HH + kb;
    unsigned short* dst = &Wl[g][lr][kb];
    #pragma unroll
    for (int i = 0; i < 16; ++i)
      async_copy16(dst + i * 8, src + i * 8);
    wait_async();
  }
  __syncthreads();

  for (int t = 0; t < TT; ++t) {
    // Issue the xg gather first so it overlaps the spin + WMMA K-loop.
    float xv[8];
    #pragma unroll
    for (int v = 0; v < 8; ++v) {
      const int b = bm * 16 + v + half * 8;
      xv[v] = xg[((long)b * TT + t) * G4H + g * HH + hn * 16 + lr];
    }

    v8f acc = {};
    if (t > 0) {                              // h(-1) == 0 -> skip GEMM at t==0
      // Wait until all 16 hidden-tile WGs of this batch group produced h(t-1).
      if (threadIdx.x == 0) {
        volatile int* f = flags + bm * TT + (t - 1);
        while (*f < 16) __builtin_amdgcn_s_sleep(2);
      }
      __syncthreads();
      __threadfence();                        // acquire h(t-1)

      const unsigned short* hrow =
          hseq + ((long)(bm * 16 + lr) * TT + (t - 1)) * HH + half * 8;
      Frag16 a[2];
      a[0].u[0] = *(const uint4*)(hrow);
      a[0].u[1] = *(const uint4*)(hrow + 16);
      #pragma unroll
      for (int kc = 0; kc < 8; ++kc) {
        const int cur = kc & 1;
        if (kc < 7) {                         // prefetch next A chunk
          const int k1 = (kc + 1) * 32;
          a[cur ^ 1].u[0] = *(const uint4*)(hrow + k1);
          a[cur ^ 1].u[1] = *(const uint4*)(hrow + k1 + 16);
        }
        Frag16 bfr;                           // B from LDS (padded rows)
        const int k0 = kc * 32;
        bfr.u[0] = *(const uint4*)&Wl[g][lr][k0 + half * 8];
        bfr.u[1] = *(const uint4*)&Wl[g][lr][k0 + 16 + half * 8];
        acc = __builtin_amdgcn_wmma_f32_16x16x32_bf16(
            false, a[cur].v, false, bfr.v, (short)0, acc, false, false);
      }
    }

    // gates tile -> LDS (adding the precomputed input projection)
    #pragma unroll
    for (int v = 0; v < 8; ++v)
      gbuf[g][v + half * 8][lr] = acc[v] + xv[v];
    __syncthreads();

    // Elementwise LSTM cell update: 256 tile elements, 2 per thread.
    #pragma unroll
    for (int e2 = 0; e2 < 2; ++e2) {
      const int e = threadIdx.x * 2 + e2;
      const int m = e >> 4, n = e & 15;
      const int b = bm * 16 + m, hc = hn * 16 + n;
      const float gi = gbuf[0][m][n];
      const float gf = gbuf[1][m][n];
      const float gg = gbuf[2][m][n];
      const float go = gbuf[3][m][n];
      const float cp = (t == 0) ? 0.f : cbuf[b * HH + hc];
      const float c  = sigm(gf) * cp + sigm(gi) * tanhf(gg);
      const float h  = sigm(go) * tanhf(c);
      cbuf[b * HH + hc] = c;
      hseq[((long)b * TT + t) * HH + hc] = f2bf(h);
    }

    __threadfence();                          // release h(t)
    __syncthreads();                          // whole WG done (also guards gbuf)
    if (threadIdx.x == 0)
      atomicAdd(flags + bm * TT + t, 1);      // announce this hidden tile
  }
}

// ---------------------------------------------------------------------------
// Output head: logits[b,t,o] = sigmoid(sum_h y[b,t,h]*W_out[o,h] + b_out[o])
// ---------------------------------------------------------------------------
__global__ void k_head(const unsigned short* __restrict__ hseq,   // [B*T, H] bf16
                       const float* __restrict__ Wout,            // [O, H]
                       const float* __restrict__ bout,            // [O]
                       float* __restrict__ out)                   // [B*T, O]
{
  const int m = blockIdx.x * blockDim.x + threadIdx.x;
  if (m >= BB * TT) return;
  const unsigned short* hr = hseq + (long)m * HH;
  float a0 = 0.f, a1 = 0.f;
  for (int k = 0; k < HH; ++k) {
    const float hv = bf2f(hr[k]);
    a0 += hv * Wout[k];
    a1 += hv * Wout[HH + k];
  }
  out[m * OO + 0] = sigm(a0 + bout[0]);
  out[m * OO + 1] = sigm(a1 + bout[1]);
}

// ---------------------------------------------------------------------------
extern "C" void kernel_launch(void* const* d_in, const int* in_sizes, int n_in,
                              void* d_out, int out_size, void* d_ws, size_t ws_size,
                              hipStream_t stream) {
  (void)in_sizes; (void)n_in; (void)out_size; (void)ws_size;
  const float* x    = (const float*)d_in[0];   // [B,T,D]
  const float* Wih  = (const float*)d_in[1];   // [L,4H,H]
  const float* Whh  = (const float*)d_in[2];   // [L,4H,H]
  const float* bih  = (const float*)d_in[3];   // [L,4H]
  const float* bhh  = (const float*)d_in[4];   // [L,4H]
  const float* Wout = (const float*)d_in[5];   // [O,H]
  const float* bout = (const float*)d_in[6];   // [O]
  float* out = (float*)d_out;

  char* ws = (char*)d_ws;
  size_t off = 0;
  auto alloc = [&](size_t bytes) -> char* {
    char* p = ws + off;
    off += (bytes + 255) & ~(size_t)255;
    return p;
  };

  float*          xgbuf = (float*)alloc((size_t)BB * TT * G4H * sizeof(float));
  unsigned short* hbuf0 = (unsigned short*)alloc((size_t)BB * TT * HH * 2);
  unsigned short* hbuf1 = (unsigned short*)alloc((size_t)BB * TT * HH * 2);
  unsigned short* wihb  = (unsigned short*)alloc((size_t)LL * G4H * HH * 2);
  unsigned short* whhb  = (unsigned short*)alloc((size_t)LL * G4H * HH * 2);
  float*          cbuf  = (float*)alloc((size_t)BB * HH * sizeof(float));
  int*            flags = (int*)alloc(4 * TT * sizeof(int));

  // One-time precision conversion (bf16 weights: 52MB, fully L2-resident).
  const long nx = (long)BB * TT * DD;
  const long nw = (long)LL * G4H * HH;
  k_f32_to_bf16<<<512, 256, 0, stream>>>(x, hbuf0, nx);
  k_f32_to_bf16<<<2048, 256, 0, stream>>>(Wih, wihb, nw);
  k_f32_to_bf16<<<2048, 256, 0, stream>>>(Whh, whhb, nw);

  unsigned short* bufs[2] = { hbuf0, hbuf1 };
  for (int l = 0; l < LL; ++l) {
    const unsigned short* hin  = bufs[l & 1];
    unsigned short*       hout = bufs[(l + 1) & 1];
    k_xproj<<<dim3(16, 256), 128, 0, stream>>>(
        hin, wihb + (long)l * G4H * HH, bih + l * G4H, bhh + l * G4H, xgbuf);
    (void)hipMemsetAsync(flags, 0, 4 * TT * sizeof(int), stream);
    k_scan<<<64, 128, 0, stream>>>(
        xgbuf, whhb + (long)l * G4H * HH, hout, cbuf, flags);
  }

  // LL is even -> final hidden sequence lives in bufs[0].
  k_head<<<(BB * TT + 127) / 128, 128, 0, stream>>>(bufs[0], Wout, bout, out);
}
